// NoteAxis_8727373545749
// MI455X (gfx1250) — matrix operand order repaired and checked by reading, not compiled, not run on hardware
//
#include <hip/hip_runtime.h>
#include <math.h>

typedef __attribute__((ext_vector_type(16))) __bf16 v16bf;
typedef __attribute__((ext_vector_type(8)))  float  v8f;

#define BATCH   4096
#define STEPS   48
#define NU      256      // hidden units
#define TILE_B  64       // batch rows per workgroup
#define NWG     (BATCH / TILE_B)
#define K0P     288      // layer-0 input K padded: 256 feat + 3 notes + 29 zero
#define XS_STRIDE 296
#define HS_STRIDE 264
#define SA_STRIDE 296
#define SB_STRIDE 264

__device__ __forceinline__ float sigf(float x) { return 1.0f / (1.0f + __expf(-x)); }

__device__ __forceinline__ float tanh_fast(float x) {
#if __has_builtin(__builtin_amdgcn_tanhf)
  return __builtin_amdgcn_tanhf(x);      // gfx1250 v_tanh_f32
#else
  return tanhf(x);
#endif
}

// CDNA5 async global->LDS DMA (tracked by ASYNCcnt). Each lane copies 16B from
// its global address to its LDS offset. LDS offset = low 32 bits of flat ptr.
__device__ __forceinline__ void async_ld_b128(void* smem_dst, const void* gmem_src) {
  unsigned loff = (unsigned)(unsigned long long)smem_dst;
  asm volatile("global_load_async_to_lds_b128 %0, %1, off"
               :: "v"(loff), "v"(gmem_src)
               : "memory");
}
__device__ __forceinline__ void wait_asynccnt0() {
  asm volatile("s_wait_asynccnt 0" ::: "memory");
}

// A fragment: 16x32 bf16, M rows striped lanes 0-15 / 16-31 per ISA layout.
__device__ __forceinline__ v16bf load_a_frag(const __bf16* p, int stride, int mbase,
                                             int kbase, int lane) {
  const int m = mbase + (lane & 15);
  const int k = kbase + ((lane >> 4) << 3);
  const __bf16* q = p + m * stride + k;
  union { v16bf v; struct { float4 lo, hi; } f; } u;
  u.f.lo = *reinterpret_cast<const float4*>(q);
  u.f.hi = *reinterpret_cast<const float4*>(q + 16);
  return u.v;
}

// B fragment: 32x16 bf16 (K x N). lane L: N = L&15, K = kbase + (L>=16 ? 8 : 0)
// + {0..7, 16..23}. Weights stored [n][k] row-major -> contiguous.
__device__ __forceinline__ v16bf load_b_frag(const __bf16* p, int stride,
                                             int kbase, int lane) {
  const int n = lane & 15;
  const int k = kbase + ((lane >> 4) << 3);
  const __bf16* q = p + n * stride + k;
  union { v16bf v; struct { float4 lo, hi; } f; } u;
  u.f.lo = *reinterpret_cast<const float4*>(q);
  u.f.hi = *reinterpret_cast<const float4*>(q + 16);
  return u.v;
}

// Issue async DMA for one W_ih stage chunk (utiles {ut, ut+8}, 4 gates).
__device__ __forceinline__ void issue_A(const __bf16* __restrict__ wA, int KA, int nchA,
                                        int ut, int tid, __bf16* sAb) {
  for (int i = tid; i < 2 * 4 * 16 * nchA; i += 256) {
    const int c = i % nchA, r = (i / nchA) & 15, g = (i / (nchA * 16)) & 3,
              uh = i / (nchA * 64);
    async_ld_b128(sAb + (size_t)(uh * 4 + g) * (16 * SA_STRIDE) + r * SA_STRIDE + c * 8,
                  wA + (size_t)(g * NU + (uh * 8 + ut) * 16 + r) * KA + c * 8);
  }
}
// Issue async DMA for one W_hh stage chunk.
__device__ __forceinline__ void issue_B(const __bf16* __restrict__ wB,
                                        int ut, int tid, __bf16* sBb) {
  for (int i = tid; i < 2 * 4 * 16 * 32; i += 256) {
    const int c = i & 31, r = (i >> 5) & 15, g = (i >> 9) & 3, uh = i >> 11;
    async_ld_b128(sBb + (size_t)(uh * 4 + g) * (16 * SB_STRIDE) + r * SB_STRIDE + c * 8,
                  wB + (size_t)(g * NU + (uh * 8 + ut) * 16 + r) * NU + c * 8);
  }
}

// ---------------- prep: f32 weights -> padded bf16, fold biases ----------------
__global__ void prep_weights(const float* __restrict__ Wih0, const float* __restrict__ Whh0,
                             const float* __restrict__ bih0, const float* __restrict__ bhh0,
                             const float* __restrict__ Wih1, const float* __restrict__ Whh1,
                             const float* __restrict__ bih1, const float* __restrict__ bhh1,
                             __bf16* __restrict__ wih0p, __bf16* __restrict__ whh0p,
                             __bf16* __restrict__ wih1p, __bf16* __restrict__ whh1p,
                             float* __restrict__ b0, float* __restrict__ b1) {
  const int r = blockIdx.x;      // gate row 0..1023
  const int t = threadIdx.x;
  for (int k = t; k < K0P; k += 256)
    wih0p[(size_t)r * K0P + k] = (k < 259) ? (__bf16)Wih0[(size_t)r * 259 + k] : (__bf16)0.0f;
  for (int k = t; k < NU; k += 256) {
    whh0p[(size_t)r * NU + k] = (__bf16)Whh0[(size_t)r * NU + k];
    wih1p[(size_t)r * NU + k] = (__bf16)Wih1[(size_t)r * NU + k];
    whh1p[(size_t)r * NU + k] = (__bf16)Whh1[(size_t)r * NU + k];
  }
  if (t == 0) { b0[r] = bih0[r] + bhh0[r]; b1[r] = bih1[r] + bhh1[r]; }
}

// ---------------- persistent 2-layer LSTM over 48 steps ----------------
__global__ __launch_bounds__(256, 1)
void lstm_persist(const float* __restrict__ nf, const float* __restrict__ cnotes,
                  const __bf16* __restrict__ wih0p, const __bf16* __restrict__ whh0p,
                  const __bf16* __restrict__ wih1p, const __bf16* __restrict__ whh1p,
                  const float* __restrict__ b0, const float* __restrict__ b1,
                  const float* __restrict__ Wout, const float* __restrict__ bout,
                  float* __restrict__ cbuf, float* __restrict__ out) {
  __shared__ __bf16 xs[TILE_B * XS_STRIDE];        // 37,888 B  x_t tile (padded K=288)
  __shared__ __bf16 hls[4][TILE_B * HS_STRIDE];    // 135,168 B h0/h1 double-buffered
  __shared__ __bf16 sA[2][4][16 * SA_STRIDE];      // 75,776 B  W_ih chunk (phase-1 only)
  __shared__ __bf16 sB[2][4][16 * SB_STRIDE];      // 67,584 B  W_hh chunk (phase-2 only)

  const int tid   = threadIdx.x;
  const int lane  = tid & 31;
  const int wv    = tid >> 5;
  const int mtile = wv & 3;     // which 16-row batch sub-tile
  const int uhalf = wv >> 2;    // utile half (0..7 or 8..15)
  const int bbase = blockIdx.x * TILE_B;

  for (int i = tid; i < 4 * TILE_B * HS_STRIDE; i += 256)
    (&hls[0][0])[i] = (__bf16)0.0f;                // h=0 at t=0

  const v8f vzero = {0.f, 0.f, 0.f, 0.f, 0.f, 0.f, 0.f, 0.f};

  for (int step = 0; step < STEPS; ++step) {
    const int p = step & 1;
    __syncthreads();
    // build x_t = [features | shifted notes | zero pad] as bf16
    for (int i = tid; i < TILE_B * XS_STRIDE; i += 256) {
      const int row = i / XS_STRIDE, k = i - row * XS_STRIDE;
      float v = 0.0f;
      if (k < NU)            v = nf[((size_t)(bbase + row) * STEPS + step) * NU + k];
      else if (k < 259 && step > 0)
                             v = cnotes[((size_t)(bbase + row) * STEPS + (step - 1)) * 3 + (k - NU)];
      xs[i] = (__bf16)v;
    }

    for (int l = 0; l < 2; ++l) {
      const __bf16* wA  = l ? wih1p : wih0p;
      const __bf16* wB  = l ? whh1p : whh0p;
      const float*  bl  = l ? b1 : b0;
      const int KA      = l ? NU : K0P;
      const int nktA    = KA >> 5;
      const int nchA    = KA >> 3;                 // 16B chunks per weight row
      const __bf16* aSrc = l ? hls[1 - p] : xs;    // layer1 input = fresh h0
      const int aStride  = l ? HS_STRIDE : XS_STRIDE;
      const __bf16* hOld = hls[2 * l + p];
      __bf16*       hNew = hls[2 * l + 1 - p];
      float*        cl   = cbuf + (size_t)l * BATCH * NU;

      // prologue: kick off DMA for ut=0 (previous phase consumers all retired)
      issue_A(wA, KA, nchA, 0, tid, &sA[0][0][0]);
      issue_B(wB, 0, tid, &sB[0][0][0]);

      for (int ut = 0; ut < 8; ++ut) {
        wait_asynccnt0();          // this wave's A[ut] + B[ut] landed in LDS
        __syncthreads();           // ... and everyone else's

        const int utile = uhalf * 8 + ut;
        v8f acc[4];
#pragma unroll
        for (int g = 0; g < 4; ++g) acc[g] = vzero;

        // phase 1: gates += x_t (or h0_new) @ W_ih^T   (reads sA only)
        for (int kt = 0; kt < nktA; ++kt) {
          const v16bf a = load_a_frag(aSrc, aStride, mtile * 16, kt * 32, lane);
#pragma unroll
          for (int g = 0; g < 4; ++g) {
            const v16bf b = load_b_frag(&sA[uhalf][g][0], SA_STRIDE, kt * 32, lane);
            acc[g] = __builtin_amdgcn_wmma_f32_16x16x32_bf16(false, a, false, b,
                                                             (short)0, acc[g], false, false);
          }
        }
        __syncthreads();           // all waves finished reading sA[ut]
        if (ut < 7)                // prefetch next W_ih chunk; DMA overlaps phase 2
          issue_A(wA, KA, nchA, ut + 1, tid, &sA[0][0][0]);

        // phase 2: gates += h_{t-1} @ W_hh^T   (reads sB only)
        for (int kt = 0; kt < 8; ++kt) {
          const v16bf a = load_a_frag(hOld, HS_STRIDE, mtile * 16, kt * 32, lane);
#pragma unroll
          for (int g = 0; g < 4; ++g) {
            const v16bf b = load_b_frag(&sB[uhalf][g][0], SB_STRIDE, kt * 32, lane);
            acc[g] = __builtin_amdgcn_wmma_f32_16x16x32_bf16(false, a, false, b,
                                                             (short)0, acc[g], false, false);
          }
        }

        // LSTM cell in WMMA D-register layout: M = r + (lane>=16)*8, N = lane&15
        const int col  = lane & 15;
        const int ucol = utile * 16 + col;
        const float bi_ = bl[ucol],          bf_ = bl[NU + ucol],
                    bg_ = bl[2 * NU + ucol], bo_ = bl[3 * NU + ucol];
        const int rbase = mtile * 16 + ((lane >> 4) << 3);
        float*  crow = cl + (size_t)(bbase + rbase) * NU + ucol;
        __bf16* hrow = hNew + rbase * HS_STRIDE + utile * 16 + col;
#pragma unroll
        for (int r = 0; r < 8; ++r) {
          const float iv = sigf(acc[0][r] + bi_);
          const float fv = sigf(acc[1][r] + bf_);
          const float gv = tanh_fast(acc[2][r] + bg_);
          const float ov = sigf(acc[3][r] + bo_);
          const float cp = (step == 0) ? 0.0f : crow[(size_t)r * NU];
          const float cn = fv * cp + iv * gv;
          crow[(size_t)r * NU] = cn;
          hrow[r * HS_STRIDE]  = (__bf16)(ov * tanh_fast(cn));
        }

        __syncthreads();           // all waves finished reading sB[ut]
        if (ut < 7)
          issue_B(wB, ut + 1, tid, &sB[0][0][0]);
      }
    }

    __syncthreads();
    // output projection: out[b,step,:] = h1 @ W_out^T + b_out, sigmoid on j<2
    const __bf16* h1n = hls[3 - p];
    for (int idx = tid; idx < TILE_B * 3; idx += 256) {
      const int row = idx / 3, j = idx - row * 3;
      float s = bout[j];
      const __bf16* hp = h1n + row * HS_STRIDE;
      const float*  wp = Wout + j * NU;
      for (int k = 0; k < NU; ++k) s += (float)hp[k] * wp[k];
      if (j < 2) s = sigf(s);
      out[((size_t)(bbase + row) * STEPS + step) * 3 + j] = s;
    }
  }
}

extern "C" void kernel_launch(void* const* d_in, const int* in_sizes, int n_in,
                              void* d_out, int out_size, void* d_ws, size_t ws_size,
                              hipStream_t stream) {
  const float* nf    = (const float*)d_in[0];
  const float* cn    = (const float*)d_in[1];
  const float* Wih0  = (const float*)d_in[2];
  const float* Whh0  = (const float*)d_in[3];
  const float* bih0  = (const float*)d_in[4];
  const float* bhh0  = (const float*)d_in[5];
  const float* Wih1  = (const float*)d_in[6];
  const float* Whh1  = (const float*)d_in[7];
  const float* bih1  = (const float*)d_in[8];
  const float* bhh1  = (const float*)d_in[9];
  const float* Wout  = (const float*)d_in[10];
  const float* bout  = (const float*)d_in[11];
  float* out = (float*)d_out;

  char* ws = (char*)d_ws;
  size_t off = 0;
  __bf16* wih0p = (__bf16*)(ws + off); off += (size_t)1024 * K0P * 2;   // 589,824
  __bf16* whh0p = (__bf16*)(ws + off); off += (size_t)1024 * NU  * 2;   // 524,288
  __bf16* wih1p = (__bf16*)(ws + off); off += (size_t)1024 * NU  * 2;
  __bf16* whh1p = (__bf16*)(ws + off); off += (size_t)1024 * NU  * 2;
  float*  b0    = (float*)(ws + off);  off += 1024 * sizeof(float);
  float*  b1    = (float*)(ws + off);  off += 1024 * sizeof(float);
  float*  cbuf  = (float*)(ws + off);  // 2 * 4096 * 256 f32 = 8 MB (L2 resident)

  prep_weights<<<1024, 256, 0, stream>>>(Wih0, Whh0, bih0, bhh0, Wih1, Whh1, bih1, bhh1,
                                         wih0p, whh0p, wih1p, whh1p, b0, b1);
  lstm_persist<<<NWG, 256, 0, stream>>>(nf, cn, wih0p, whh0p, wih1p, whh1p,
                                        b0, b1, Wout, bout, cbuf, out);
}
